// VectorQuantizerEMA_33174327394965
// MI455X (gfx1250) — matrix operand order repaired and checked by reading, not compiled, not run on hardware
//
#include <hip/hip_runtime.h>

typedef __attribute__((ext_vector_type(16))) __bf16 v16bf;
typedef __attribute__((ext_vector_type(8)))  float  v8f;

#define K_CODES 2048
#define DIM     256
#define N_ROWS  32768
#define DECAYF      0.99f
#define ONE_M_DECAY 0.01f
#define EPSF        1e-5f

// ---- workspace layout (byte offsets) ----
#define WS_IDX     0u          // int[32768]
#define WS_CNT     131072u     // float[2048]
#define WS_NW      139264u     // float[524288]
#define WS_ENORM   2236416u    // float[2048]
#define WS_RAW     2244608u    // float[2048]
#define WS_LOSS    2252800u    // float
#define WS_TOTAL   2252804u    // float
#define WS_EBF     2252928u    // bf16[524288] = 1 MB pre-converted codebook

// ---- output layout (float element offsets, tuple order) ----
#define O_ZQ   ((size_t)0)
#define O_IDX  ((size_t)8388608)
#define O_CB   ((size_t)8421376)
#define O_CM   ((size_t)8421377)
#define O_EMB  ((size_t)8421378)
#define O_CNT  ((size_t)8945666)
#define O_EW   ((size_t)8947714)

__device__ __forceinline__ unsigned short f2bf(float f) {
    unsigned int u = __builtin_bit_cast(unsigned int, f);
    u += 0x7FFFu + ((u >> 16) & 1u);          // round-to-nearest-even
    return (unsigned short)(u >> 16);
}
__device__ __forceinline__ __bf16 bfbits(unsigned short s) {
    union { unsigned short u; __bf16 b; } x; x.u = s; return x.b;
}
template <int BASE>
__device__ __forceinline__ void pack8(v16bf& a, float4 u, float4 v) {
    a[BASE + 0] = bfbits(f2bf(u.x)); a[BASE + 1] = bfbits(f2bf(u.y));
    a[BASE + 2] = bfbits(f2bf(u.z)); a[BASE + 3] = bfbits(f2bf(u.w));
    a[BASE + 4] = bfbits(f2bf(v.x)); a[BASE + 5] = bfbits(f2bf(v.y));
    a[BASE + 6] = bfbits(f2bf(v.z)); a[BASE + 7] = bfbits(f2bf(v.w));
}

// ---------------- Kernel 0a: codebook fp32 -> bf16 (one-shot) ----------------
__global__ void convert_kernel(const float* __restrict__ emb,
                               unsigned short* __restrict__ ebf) {
    int t = blockIdx.x * blockDim.x + threadIdx.x;    // one float4 per thread
    float4 v = ((const float4*)emb)[t];
    ushort4 s;
    s.x = f2bf(v.x); s.y = f2bf(v.y); s.z = f2bf(v.z); s.w = f2bf(v.w);
    ((ushort4*)ebf)[t] = s;
}

// ---------------- Kernel 0b: ||e||^2 per code ----------------
__global__ void enorm_kernel(const float* __restrict__ emb, float* __restrict__ enorm) {
    int k = blockIdx.x * blockDim.x + threadIdx.x;
    if (k >= K_CODES) return;
    const float4* p = (const float4*)(emb + (size_t)k * DIM);
    float s = 0.f;
#pragma unroll 8
    for (int i = 0; i < DIM / 4; ++i) {
        float4 v = p[i];
        s += v.x * v.x + v.y * v.y + v.z * v.z + v.w * v.w;
    }
    enorm[k] = s;
}

// ---------------- Kernel 1: WMMA distance + argmin ----------------
// grid = 256 blocks x 256 threads (8 waves). Each wave: 16 rows vs all 2048 codes.
__global__ void __launch_bounds__(256) argmin_kernel(
    const float* __restrict__ ze, const unsigned short* __restrict__ ebf,
    const float* __restrict__ enorm, int* __restrict__ out_idx) {

    __shared__ unsigned short sE[128 * DIM];  // 64 KB: 128 codes x 256 dims, bf16

    const int lane = threadIdx.x & 31;
    const int wave = threadIdx.x >> 5;
    const int half = lane >> 4;      // 0: lanes 0-15, 1: lanes 16-31
    const int l16  = lane & 15;
    const int rowBase = (blockIdx.x * 8 + wave) * 16;

    // --- preload & pack A fragments: 16 rows x 256 dims of z_e as bf16 ---
    // A 16x32 fragment, contiguous k order (same as B): lane row = l16; elems 0-15 <- K = kb..kb+15,
    // kb = ds*32 + half*16.
    v16bf afrag[8];
    {
        const float* rp = ze + (size_t)(rowBase + l16) * DIM;
#pragma unroll
        for (int ds = 0; ds < 8; ++ds) {
            const int kb = ds * 32 + half * 16;
            float4 x0 = *(const float4*)(rp + kb);
            float4 x1 = *(const float4*)(rp + kb + 4);
            float4 x2 = *(const float4*)(rp + kb + 8);
            float4 x3 = *(const float4*)(rp + kb + 12);
            v16bf a;
            pack8<0>(a, x0, x1);
            pack8<8>(a, x2, x3);
            afrag[ds] = a;
        }
    }

    float best[8];
    int   bidx[8];
#pragma unroll
    for (int j = 0; j < 8; ++j) { best[j] = 3.4e38f; bidx[j] = 0; }

    const unsigned int ldsbase = (unsigned int)(size_t)(&sE[0]);

    for (int cb = 0; cb < 16; ++cb) {
        __syncthreads();   // all waves done reading previous block
        // --- async stage: 64 KB of bf16 codebook block, global -> LDS ---
        {
            const char* gsrc = (const char*)(ebf + (size_t)cb * 128 * DIM);
#pragma unroll
            for (int i = 0; i < 16; ++i) {
                const int off = i * 4096 + (int)threadIdx.x * 16;   // bytes
                const void* g = gsrc + off;
                unsigned int l = ldsbase + (unsigned int)off;
                asm volatile("global_load_async_to_lds_b128 %0, %1, off"
                             :: "v"(l), "v"(g) : "memory");
            }
        }
        if (cb < 15)  // warm next bf16 block in L2 (global_prefetch_b8)
            __builtin_prefetch((const void*)((const char*)ebf + (size_t)(cb + 1) * 65536 +
                                             threadIdx.x * 16), 0, 0);
        asm volatile("s_wait_asynccnt 0x0" ::: "memory");
        __syncthreads();

        // two independent accumulators + software-pipelined B fragments
        for (int ct = 0; ct < 8; ct += 2) {
            v8f c0 = {0, 0, 0, 0, 0, 0, 0, 0};
            v8f c1 = {0, 0, 0, 0, 0, 0, 0, 0};
            // B 32x16 fragment, contiguous k order (same as A): lane col = l16; elems 0-15 <- K = half*16 + e
            const unsigned short* b0p = &sE[(size_t)(ct * 16 + l16) * DIM + half * 16];
            const unsigned short* b1p = b0p + 16 * DIM;
            union BU { uint4 q[2]; v16bf v; };
            BU cur0, cur1, nxt0, nxt1;
            {
                const uint4* q0 = (const uint4*)(b0p);
                const uint4* q1 = (const uint4*)(b1p);
                cur0.q[0] = q0[0]; cur0.q[1] = q0[1];
                cur1.q[0] = q1[0]; cur1.q[1] = q1[1];
            }
#pragma unroll
            for (int ds = 0; ds < 8; ++ds) {
                if (ds < 7) {   // prefetch next d-step fragments before issuing WMMA
                    const uint4* q0 = (const uint4*)(b0p + (ds + 1) * 32);
                    const uint4* q1 = (const uint4*)(b1p + (ds + 1) * 32);
                    nxt0.q[0] = q0[0]; nxt0.q[1] = q0[1];
                    nxt1.q[0] = q1[0]; nxt1.q[1] = q1[1];
                }
                c0 = __builtin_amdgcn_wmma_f32_16x16x32_bf16(
                         false, afrag[ds], false, cur0.v, (short)0, c0, false, false);
                c1 = __builtin_amdgcn_wmma_f32_16x16x32_bf16(
                         false, afrag[ds], false, cur1.v, (short)0, c1, false, false);
                if (ds < 7) { cur0 = nxt0; cur1 = nxt1; }
            }
            const int code0 = cb * 128 + ct * 16 + l16;
            const int code1 = code0 + 16;
            const float en0 = enorm[code0];
            const float en1 = enorm[code1];
            // C layout: VGPR j, this lane holds (row = half*8 + j, col = l16)
#pragma unroll
            for (int j = 0; j < 8; ++j) {
                float d0 = en0 - 2.0f * c0[j];
                if (d0 < best[j]) { best[j] = d0; bidx[j] = code0; }
                float d1 = en1 - 2.0f * c1[j];
                if (d1 < best[j]) { best[j] = d1; bidx[j] = code1; }
            }
        }
    }

    // min-reduce across the 16 lanes holding the same rows (xor stays in half)
#pragma unroll
    for (int off = 1; off <= 8; off <<= 1) {
#pragma unroll
        for (int j = 0; j < 8; ++j) {
            float ov = __shfl_xor(best[j], off, 32);
            int   oi = __shfl_xor(bidx[j], off, 32);
            if (ov < best[j] || (ov == best[j] && oi < bidx[j])) {
                best[j] = ov; bidx[j] = oi;
            }
        }
    }
    if (l16 == 0) {
#pragma unroll
        for (int j = 0; j < 8; ++j)
            out_idx[rowBase + half * 8 + j] = bidx[j];
    }
}

// ---------------- Kernel 2: gather z_q, loss reduce, scatter-add ----------------
__global__ void __launch_bounds__(256) gather_kernel(
    const float* __restrict__ ze, const float* __restrict__ emb,
    const int* __restrict__ idx, float* __restrict__ out,
    float* __restrict__ nw, float* __restrict__ cnt, float* __restrict__ lossAcc) {
    __shared__ float sred[8];
    const int row  = blockIdx.x;
    const int t    = threadIdx.x;
    const int code = idx[row];
    const float e = emb[(size_t)code * DIM + t];
    const float x = ze[(size_t)row * DIM + t];
    out[O_ZQ + (size_t)row * DIM + t] = e;    // z_q_st == z_q numerically
    float d  = x - e;
    float d2 = d * d;
#pragma unroll
    for (int off = 16; off; off >>= 1) d2 += __shfl_xor(d2, off, 32);
    if ((t & 31) == 0) sred[t >> 5] = d2;
    atomicAdd(&nw[(size_t)code * DIM + t], x);
    __syncthreads();
    if (t == 0) {
        float s = 0.f;
#pragma unroll
        for (int i = 0; i < 8; ++i) s += sred[i];
        atomicAdd(lossAcc, s);
        atomicAdd(&cnt[code], 1.0f);
        out[O_IDX + row] = (float)code;
    }
}

// ---------------- Kernel 3: EMA count + global total ----------------
__global__ void countema_kernel(const float* __restrict__ ema_count,
                                const float* __restrict__ cnt,
                                float* __restrict__ raw, float* __restrict__ total) {
    int k = blockIdx.x * blockDim.x + threadIdx.x;
    float r = ema_count[k] * DECAYF + ONE_M_DECAY * cnt[k];
    raw[k] = r;
#pragma unroll
    for (int off = 16; off; off >>= 1) r += __shfl_xor(r, off, 32);
    if ((threadIdx.x & 31) == 0) atomicAdd(total, r);
}

// ---------------- Kernel 4: smoothing, EMA weight, new embedding, losses ----------------
__global__ void finalize_kernel(const float* __restrict__ ema_weight,
                                const float* __restrict__ nw,
                                const float* __restrict__ raw,
                                const float* __restrict__ total,
                                const float* __restrict__ lossAcc,
                                float* __restrict__ out) {
    const int k = blockIdx.x;
    const int d = threadIdx.x;
    const float tot = *total;
    const float c = (raw[k] + EPSF) / (tot + (float)K_CODES * EPSF) * tot;
    const size_t o = (size_t)k * DIM + d;
    const float wn = ema_weight[o] * DECAYF + ONE_M_DECAY * nw[o];
    out[O_EW + o]  = wn;
    out[O_EMB + o] = wn / c;
    if (d == 0) out[O_CNT + k] = c;
    if (k == 0 && d == 0) {
        float loss = *lossAcc / (float)((size_t)N_ROWS * DIM);
        out[O_CB] = loss;   // codebook_loss == commitment_loss numerically
        out[O_CM] = loss;
    }
}

extern "C" void kernel_launch(void* const* d_in, const int* in_sizes, int n_in,
                              void* d_out, int out_size, void* d_ws, size_t ws_size,
                              hipStream_t stream) {
    const float* ze         = (const float*)d_in[0];
    const float* emb        = (const float*)d_in[1];
    const float* ema_count  = (const float*)d_in[2];
    const float* ema_weight = (const float*)d_in[3];
    float* out = (float*)d_out;
    char*  ws  = (char*)d_ws;

    int*            wIdx  = (int*)(ws + WS_IDX);
    float*          wCnt  = (float*)(ws + WS_CNT);
    float*          wNW   = (float*)(ws + WS_NW);
    float*          wEN   = (float*)(ws + WS_ENORM);
    float*          wRaw  = (float*)(ws + WS_RAW);
    float*          wLoss = (float*)(ws + WS_LOSS);
    float*          wTot  = (float*)(ws + WS_TOTAL);
    unsigned short* wEbf  = (unsigned short*)(ws + WS_EBF);

    // zero accumulators (graph-capture-safe)
    hipMemsetAsync(ws + WS_CNT, 0, WS_ENORM - WS_CNT, stream);   // cnt + new_weight
    hipMemsetAsync(ws + WS_LOSS, 0, 8, stream);                  // loss + total

    convert_kernel<<<(K_CODES * DIM / 4) / 256, 256, 0, stream>>>(emb, wEbf);
    enorm_kernel  <<<K_CODES / 256, 256, 0, stream>>>(emb, wEN);
    argmin_kernel <<<N_ROWS / 128, 256, 0, stream>>>(ze, wEbf, wEN, wIdx);
    gather_kernel <<<N_ROWS, 256, 0, stream>>>(ze, emb, wIdx, out, wNW, wCnt, wLoss);
    countema_kernel<<<K_CODES / 256, 256, 0, stream>>>(ema_count, wCnt, wRaw, wTot);
    finalize_kernel<<<K_CODES, DIM, 0, stream>>>(ema_weight, wNW, wRaw, wTot, wLoss, out);
}
